// GatedDeltaNet_82858509075100
// MI455X (gfx1250) — matrix operand order repaired and checked by reading, not compiled, not run on hardware
//
#include <hip/hip_runtime.h>
#include <hip/hip_bf16.h>

// ---------------------------------------------------------------------------
// Gated DeltaNet forward for MI455X (gfx1250, wave32, WMMA, async-to-LDS)
// ---------------------------------------------------------------------------
#define B_    2
#define L_    2048
#define DIM_  2048
#define HK_   16
#define HV_   32
#define DK_   128
#define DV_   128
#define KCONV_ 4
#define KEYDIM_  (HK_ * DK_)            // 2048
#define VALDIM_  (HV_ * DV_)            // 4096
#define CONVDIM_ (2 * KEYDIM_ + VALDIM_) // 8192
#define ROWS_ (B_ * L_)                 // 4096
#define C64 64                          // delta-rule chunk length

typedef __attribute__((ext_vector_type(16))) __bf16 v16bf;
typedef __attribute__((ext_vector_type(8)))  float  v8f;

__device__ __forceinline__ v8f v8f_zero() {
  v8f z;
#pragma unroll
  for (int i = 0; i < 8; ++i) z[i] = 0.f;
  return z;
}

__device__ __forceinline__ v8f wmma_bf16(v16bf a, v16bf b, v8f c) {
  return __builtin_amdgcn_wmma_f32_16x16x32_bf16(false, a, false, b,
                                                 (short)0, c, false, false);
}

// ---- CDNA5 async global->LDS copy (16B per lane), ASYNCcnt-tracked --------
__device__ __forceinline__ void async_cp16(void* lds_dst, const void* gsrc) {
  unsigned l = (unsigned)(unsigned long long)lds_dst; // LDS byte offset
  unsigned long long g = (unsigned long long)gsrc;
  asm volatile("global_load_async_to_lds_b128 %0, %1, off"
               :: "v"(l), "v"(g) : "memory");
}
__device__ __forceinline__ void wait_async5() {
  asm volatile("s_wait_asynccnt 0x5" ::: "memory");
}
__device__ __forceinline__ void wait_async0() {
  asm volatile("s_wait_asynccnt 0x0" ::: "memory");
}

// ---- WMMA fragment loads: A: contiguous 16B LDS reads, B: strided by ld --------------------
// A (16x32 bf16): lane<16: row m=lane, K={0..7,16..23};
//                 lane>=16: row m=lane-16, K={8..15,24..31}
__device__ __forceinline__ v16bf frag_a_bf(const __bf16* p, int ld,
                                           int row0, int k0, int lane) {
  int m = lane & 15, kb = (lane >> 4) * 8;
  const __bf16* r = p + (row0 + m) * ld + k0 + kb;
  v16bf f;
#pragma unroll
  for (int e = 0; e < 8; ++e) { f[e] = r[e]; f[e + 8] = r[16 + e]; }
  return f;
}

// B (32x16 bf16): lane -> N=lane&15; lane<16: K={0..7,16..23},
//                 lane>=16: K={8..15,24..31} (same as A).
// Source layout [k][n], stride ld.
__device__ __forceinline__ v16bf frag_b_bf(const __bf16* p, int ld,
                                           int k0, int n0, int lane) {
  int n = lane & 15, kb = (lane >> 4) * 8;
  const __bf16* r = p + (k0 + kb) * ld + n0 + n;
  v16bf f;
#pragma unroll
  for (int e = 0; e < 8; ++e) { f[e] = r[e * ld]; f[e + 8] = r[(16 + e) * ld]; }
  return f;
}

// C/D (16x16 f32, 8 VGPRs): VGPR r -> (m = r + 8*(lane>>4), n = lane&15)
__device__ __forceinline__ void store_c_f32(float* p, int ld, int m0, int n0,
                                            int lane, v8f acc) {
  int n = lane & 15, hi = lane >> 4;
#pragma unroll
  for (int r = 0; r < 8; ++r) p[(m0 + r + 8 * hi) * ld + n0 + n] = acc[r];
}

// ---------------------------------------------------------------------------
// fp32 -> bf16 (straight copy, for activations)
// ---------------------------------------------------------------------------
__global__ void cvt_bf16_kernel(const float* __restrict__ in,
                                __bf16* __restrict__ out, size_t n) {
  size_t i = (size_t)blockIdx.x * blockDim.x + threadIdx.x;
  if (i < n) out[i] = (__bf16)in[i];
}

// fp32 [N][K] -> bf16 [K][N] (transpose, for GEMM B operands)
__global__ __launch_bounds__(256) void cvtT_kernel(const float* __restrict__ in,
                                                   __bf16* __restrict__ out,
                                                   int N, int K) {
  __shared__ float tile[32][33];
  int kb = blockIdx.x * 32;
  int nb = blockIdx.y * 32;
  int tx = threadIdx.x & 31;
  int ty = threadIdx.x >> 5;            // 0..7
#pragma unroll
  for (int i = 0; i < 4; ++i) {
    int n = nb + ty + 8 * i;
    tile[ty + 8 * i][tx] = (n < N) ? in[(size_t)n * K + kb + tx] : 0.f;
  }
  __syncthreads();
#pragma unroll
  for (int i = 0; i < 4; ++i) {
    int k = kb + ty + 8 * i;
    int n = nb + tx;
    if (n < N) out[(size_t)k * N + n] = (__bf16)tile[tx][ty + 8 * i];
  }
}

// ---------------------------------------------------------------------------
// GEMM: C[M,N](f32) = A[M,K](bf16, row-major) @ Wt[K,N](bf16)
// Block 256 thr = 8 waves; tile 64(M) x 256(N); wave tile 32x64; K step 32.
// Double-buffered LDS fed by async global->LDS copies.
// ---------------------------------------------------------------------------
#define GBM 64
#define GBN 256
#define GBK 32

__global__ __launch_bounds__(256) void gemm_bf16_kernel(
    const __bf16* __restrict__ A, const __bf16* __restrict__ Wt,
    float* __restrict__ Cout, int M, int N, int K) {
  __shared__ alignas(16) __bf16 sA[2][GBM][GBK];   // 2 x 4KB
  __shared__ alignas(16) __bf16 sB[2][GBK][GBN];   // 2 x 16KB
  int tid = threadIdx.x;
  int wid = tid >> 5, lane = tid & 31;
  int m0 = blockIdx.y * GBM;
  int n0 = blockIdx.x * GBN;
  int wm = (wid & 1) * 32;
  int wn = (wid >> 1) * 64;
  bool fastN = (n0 + GBN <= N);

  v8f acc[2][4];
#pragma unroll
  for (int i = 0; i < 2; ++i)
#pragma unroll
    for (int j = 0; j < 4; ++j) acc[i][j] = v8f_zero();

  auto stage = [&](int buf, int kt) {
    { // A: 64x32 bf16 = 4KB; one 16B async copy per thread
      int r = tid >> 2, c = (tid & 3) * 8;
      async_cp16(&sA[buf][r][c], A + (size_t)(m0 + r) * K + kt + c);
    }
    if (fastN) { // B: 32x256 bf16 = 16KB; 64B (4 x 16B) per thread
      int r = tid >> 3, c = (tid & 7) * 32;
      const __bf16* src = Wt + (size_t)(kt + r) * N + n0 + c;
      __bf16* dst = &sB[buf][r][c];
#pragma unroll
      for (int q = 0; q < 4; ++q) async_cp16(dst + 8 * q, src + 8 * q);
    } else {     // ragged N (tiny a/b projections): synchronous guarded fill
      int r = tid >> 3, c = (tid & 7) * 32;
#pragma unroll
      for (int e = 0; e < 32; ++e) {
        int n = n0 + c + e;
        sB[buf][r][c + e] =
            (n < N) ? Wt[(size_t)(kt + r) * N + n] : (__bf16)0.0f;
      }
    }
  };

  stage(0, 0);
  int buf = 0;
  for (int kt = 0; kt < K; kt += GBK) {
    bool has_next = (kt + GBK) < K;
    if (has_next) stage(buf ^ 1, kt + GBK);
    if (has_next) wait_async5(); else wait_async0();
    __syncthreads();

    v16bf fa[2];
#pragma unroll
    for (int i = 0; i < 2; ++i)
      fa[i] = frag_a_bf(&sA[buf][0][0], GBK, wm + i * 16, 0, lane);
#pragma unroll
    for (int j = 0; j < 4; ++j) {
      v16bf fb = frag_b_bf(&sB[buf][0][0], GBN, 0, wn + j * 16, lane);
#pragma unroll
      for (int i = 0; i < 2; ++i) acc[i][j] = wmma_bf16(fa[i], fb, acc[i][j]);
    }
    __syncthreads();
    buf ^= 1;
  }

  int nn = lane & 15, hi = lane >> 4;
#pragma unroll
  for (int i = 0; i < 2; ++i)
#pragma unroll
    for (int j = 0; j < 4; ++j)
#pragma unroll
      for (int r = 0; r < 8; ++r) {
        int m = m0 + wm + i * 16 + r + 8 * hi;
        int n = n0 + wn + j * 16 + nn;
        if (n < N) Cout[(size_t)m * N + n] = acc[i][j][r];
      }
}

// ---------------------------------------------------------------------------
// depthwise causal conv (K=4) + SiLU
// ---------------------------------------------------------------------------
__device__ __forceinline__ float conv_silu(const float* __restrict__ qkv,
                                           const float* __restrict__ w,
                                           int b, int l, int c) {
  float s = 0.f;
#pragma unroll
  for (int j = 0; j < KCONV_; ++j) {
    int lr = l + j - (KCONV_ - 1);
    if (lr >= 0)
      s += w[c * KCONV_ + j] * qkv[((size_t)(b * L_ + lr)) * CONVDIM_ + c];
  }
  return s / (1.f + __expf(-s));
}

__global__ __launch_bounds__(128) void conv_qk_kernel(
    const float* __restrict__ qkvraw, const float* __restrict__ conv_w,
    __bf16* __restrict__ qn, __bf16* __restrict__ kn) {
  int blk = blockIdx.x;
  int hk = blk % HK_;
  int l = (blk / HK_) % L_;
  int b = blk / (HK_ * L_);
  int d = threadIdx.x;

  float q = conv_silu(qkvraw, conv_w, b, l, hk * DK_ + d);
  float k = conv_silu(qkvraw, conv_w, b, l, KEYDIM_ + hk * DK_ + d);

  __shared__ float red[128];
  red[d] = q * q;
  __syncthreads();
#pragma unroll
  for (int s = 64; s > 0; s >>= 1) {
    if (d < s) red[d] += red[d + s];
    __syncthreads();
  }
  float ssq_q = red[0];
  __syncthreads();
  red[d] = k * k;
  __syncthreads();
#pragma unroll
  for (int s = 64; s > 0; s >>= 1) {
    if (d < s) red[d] += red[d + s];
    __syncthreads();
  }
  float ssq_k = red[0];

  size_t gi = (((size_t)(b * L_ + l)) * HK_ + hk) * DK_ + d;
  const float invsqrt_dk = 0.08838834764831845f; // 128^-0.5
  qn[gi] = (__bf16)(q * rsqrtf(ssq_q + 1e-6f) * invsqrt_dk);
  kn[gi] = (__bf16)(k * rsqrtf(ssq_k + 1e-6f));
}

__global__ __launch_bounds__(128) void conv_v_kernel(
    const float* __restrict__ qkvraw, const float* __restrict__ conv_w,
    float* __restrict__ vf) {
  int blk = blockIdx.x;
  int hv = blk % HV_;
  int l = (blk / HV_) % L_;
  int b = blk / (HV_ * L_);
  int d = threadIdx.x;
  float v = conv_silu(qkvraw, conv_w, b, l, 2 * KEYDIM_ + hv * DV_ + d);
  vf[(((size_t)(b * L_ + l)) * HV_ + hv) * DV_ + d] = v;
}

__global__ void gates_kernel(const float* __restrict__ af,
                             const float* __restrict__ bfp,
                             const float* __restrict__ A_log,
                             const float* __restrict__ dt_bias,
                             float* __restrict__ g, float* __restrict__ beta,
                             int total) {
  int i = blockIdx.x * blockDim.x + threadIdx.x;
  if (i >= total) return;
  int h = i & (HV_ - 1);
  float xx = af[i] + dt_bias[h];
  float sp = (xx > 20.f) ? xx : log1pf(__expf(xx));
  g[i] = -__expf(A_log[h]) * sp;
  float bv = bfp[i];
  beta[i] = 1.f / (1.f + __expf(-bv));
}

// ---------------------------------------------------------------------------
// Chunked gated delta rule (WY/UT form), one block per (b, hv) head.
// ---------------------------------------------------------------------------
__global__ __launch_bounds__(256) void delta_core_kernel(
    const __bf16* __restrict__ qn, const __bf16* __restrict__ kn,
    const float* __restrict__ vf, const float* __restrict__ g,
    const float* __restrict__ beta, float* __restrict__ of) {
  int b = blockIdx.x / HV_;
  int hv = blockIdx.x % HV_;
  int hk = hv / (HV_ / HK_);
  int tid = threadIdx.x;
  int wid = tid >> 5, lane = tid & 31;

  __shared__ alignas(16) float  S[DK_][DV_];    // 64 KB fp32 master state
  __shared__ alignas(16) __bf16 Sb[DK_][DV_];   // 32 KB bf16 shadow (B op)
  __shared__ alignas(16) __bf16 Kc[C64][DK_];   // 16 KB  [t][dk]  (A op)
  __shared__ alignas(16) __bf16 Kt[DK_ * C64];  // 16 KB  [dk][t]  (B op)
  __shared__ alignas(16) __bf16 Qc[C64][DK_];   // 16 KB  [t][dk]; reused: K2t
  __shared__ alignas(16) float  U[C64][DV_];    // 32 KB fp32 (substitution)
  __shared__ alignas(16) __bf16 Ub[C64][DV_];   // 16 KB bf16 shadow (B op)
  __shared__ alignas(16) __bf16 Pb[C64][C64];   // 8 KB   [i][j]   (A op)
  __shared__ alignas(16) float  Af[C64][C64];   // 16 KB fp32 (substitution)
  __shared__ alignas(16) float  Outf[C64][DV_]; // 32 KB (RHS0, then out)
  __shared__ float cum[C64], lam[C64], bet[C64];

  for (int i = tid; i < DK_ * DV_; i += 256) {
    (&S[0][0])[i] = 0.f;
    (&Sb[0][0])[i] = (__bf16)0.0f;
  }
  __syncthreads();

  for (int ch = 0; ch < L_ / C64; ++ch) {
    int l0 = ch * C64;

    // ---- stage chunk q,k via async global->LDS + gates ----
    for (int i = tid; i < C64 * DK_ / 8; i += 256) {
      int t = i >> 4, c8 = (i & 15) * 8;
      size_t gi = (((size_t)(b * L_ + l0 + t)) * HK_ + hk) * DK_ + c8;
      async_cp16(&Kc[t][c8], kn + gi);
      async_cp16(&Qc[t][c8], qn + gi);
    }
    if (tid < C64) {
      size_t gi = ((size_t)(b * L_ + l0 + tid)) * HV_ + hv;
      bet[tid] = beta[gi];
      cum[tid] = g[gi];
    }
    wait_async0();
    __syncthreads();
    // Kt = Kc^T
    for (int i = tid; i < C64 * DK_; i += 256) {
      int t = i >> 7, dk = i & 127;
      Kt[dk * C64 + t] = Kc[t][dk];
    }
    if (tid == 0) {
      float c = 0.f;
      for (int t = 0; t < C64; ++t) {
        c += cum[t];
        cum[t] = c;
        lam[t] = __expf(c);
      }
    }
    __syncthreads();

    // ---- RHS0 = Kc @ Sb -> Outf (64x128, 32 tiles, 4/wave) ----
    for (int tt = wid * 4; tt < wid * 4 + 4; ++tt) {
      int tm = (tt >> 3) * 16, tn = (tt & 7) * 16;
      v8f acc = v8f_zero();
#pragma unroll
      for (int k0 = 0; k0 < DK_; k0 += 32) {
        v16bf fa = frag_a_bf(&Kc[0][0], DK_, tm, k0, lane);
        v16bf fb = frag_b_bf(&Sb[0][0], DV_, k0, tn, lane);
        acc = wmma_bf16(fa, fb, acc);
      }
      store_c_f32(&Outf[0][0], DV_, tm, tn, lane, acc);
    }
    __syncthreads();

    // ---- Af = mask( beta_i e^{cum_i-cum_j} (Kc Kc^T) ), j<i ----
    for (int tt = wid * 2; tt < wid * 2 + 2; ++tt) {
      int tm = (tt >> 2) * 16, tn = (tt & 3) * 16;
      v8f acc = v8f_zero();
#pragma unroll
      for (int k0 = 0; k0 < DK_; k0 += 32) {
        v16bf fa = frag_a_bf(&Kc[0][0], DK_, tm, k0, lane);
        v16bf fb = frag_b_bf(Kt, C64, k0, tn, lane);
        acc = wmma_bf16(fa, fb, acc);
      }
      int nn = lane & 15, hi = lane >> 4;
#pragma unroll
      for (int r = 0; r < 8; ++r) {
        int i = tm + r + 8 * hi, j = tn + nn;
        Af[i][j] = (j < i) ? bet[i] * __expf(cum[i] - cum[j]) * acc[r] : 0.f;
      }
    }
    __syncthreads();

    // ---- U0 = beta * (V - lam * RHS0)  (vectorized) ----
    for (int i = tid; i < C64 * DV_ / 4; i += 256) {
      int t = i >> 5, dv = (i & 31) * 4;
      size_t gv = (((size_t)(b * L_ + l0 + t)) * HV_ + hv) * DV_ + dv;
      float4 vv = *(const float4*)(vf + gv);
      float4 rr = *(const float4*)&Outf[t][dv];
      float bt = bet[t], lm = lam[t];
      float4 u;
      u.x = bt * (vv.x - lm * rr.x);
      u.y = bt * (vv.y - lm * rr.y);
      u.z = bt * (vv.z - lm * rr.z);
      u.w = bt * (vv.w - lm * rr.w);
      *(float4*)&U[t][dv] = u;
    }
    __syncthreads();

    // ---- forward substitution: U[i] -= sum_{j<i} Af[i][j] U[j] ----
    for (int i = 1; i < C64; ++i) {
      if (tid < DV_) {
        float s = 0.f;
        for (int j = 0; j < i; ++j) s += Af[i][j] * U[j][tid];
        U[i][tid] -= s;
      }
      __syncthreads();
    }
    // Ub shadow
    for (int i = tid; i < C64 * DV_; i += 256) {
      int t = i >> 7, dv = i & 127;
      Ub[t][dv] = (__bf16)U[t][dv];
    }
    __syncthreads();

    // ---- Pb = mask( e^{cum_i-cum_j} (Qc Kc^T) ), j<=i  (bf16) ----
    for (int tt = wid * 2; tt < wid * 2 + 2; ++tt) {
      int tm = (tt >> 2) * 16, tn = (tt & 3) * 16;
      v8f acc = v8f_zero();
#pragma unroll
      for (int k0 = 0; k0 < DK_; k0 += 32) {
        v16bf fa = frag_a_bf(&Qc[0][0], DK_, tm, k0, lane);
        v16bf fb = frag_b_bf(Kt, C64, k0, tn, lane);
        acc = wmma_bf16(fa, fb, acc);
      }
      int nn = lane & 15, hi = lane >> 4;
#pragma unroll
      for (int r = 0; r < 8; ++r) {
        int i = tm + r + 8 * hi, j = tn + nn;
        Pb[i][j] =
            (j <= i) ? (__bf16)(__expf(cum[i] - cum[j]) * acc[r]) : (__bf16)0.0f;
      }
    }
    __syncthreads();

    // ---- Out = lam * (Qc @ Sb) + Pb @ Ub -> Outf ----
    for (int tt = wid * 4; tt < wid * 4 + 4; ++tt) {
      int tm = (tt >> 3) * 16, tn = (tt & 7) * 16;
      v8f a1 = v8f_zero(), a2 = v8f_zero();
#pragma unroll
      for (int k0 = 0; k0 < DK_; k0 += 32) {
        v16bf fa = frag_a_bf(&Qc[0][0], DK_, tm, k0, lane);
        v16bf fb = frag_b_bf(&Sb[0][0], DV_, k0, tn, lane);
        a1 = wmma_bf16(fa, fb, a1);
      }
#pragma unroll
      for (int k0 = 0; k0 < C64; k0 += 32) {
        v16bf fa = frag_a_bf(&Pb[0][0], C64, tm, k0, lane);
        v16bf fb = frag_b_bf(&Ub[0][0], DV_, k0, tn, lane);
        a2 = wmma_bf16(fa, fb, a2);
      }
      int nn = lane & 15, hi = lane >> 4;
#pragma unroll
      for (int r = 0; r < 8; ++r) {
        int i = tm + r + 8 * hi, dv = tn + nn;
        Outf[i][dv] = lam[i] * a1[r] + a2[r];
      }
    }
    __syncthreads();

    // ---- write out (vectorized); build K2t = (lamL/lam_t) Kc^T in Qc buf ----
    for (int i = tid; i < C64 * DV_ / 4; i += 256) {
      int t = i >> 5, dv = (i & 31) * 4;
      size_t go = (((size_t)(b * L_ + l0 + t)) * HV_ + hv) * DV_ + dv;
      *(float4*)(of + go) = *(float4*)&Outf[t][dv];
    }
    float lamL = lam[C64 - 1];
    __bf16* K2t = &Qc[0][0]; // reuse (Qc no longer needed this chunk)
    for (int i = tid; i < C64 * DK_; i += 256) {
      int t = i >> 7, dk = i & 127;
      K2t[dk * C64 + t] = (__bf16)((lamL / lam[t]) * (float)Kc[t][dk]);
    }
    __syncthreads();

    // ---- S = lamL * S + K2^T @ U  (128x128, 64 tiles, 8/wave) ----
    for (int tt = wid * 8; tt < wid * 8 + 8; ++tt) {
      int tm = (tt >> 3) * 16, tn = (tt & 7) * 16;
      v8f acc = v8f_zero();
#pragma unroll
      for (int k0 = 0; k0 < C64; k0 += 32) {
        v16bf fa = frag_a_bf(K2t, C64, tm, k0, lane); // A[dk][t]
        v16bf fb = frag_b_bf(&Ub[0][0], DV_, k0, tn, lane);
        acc = wmma_bf16(fa, fb, acc);
      }
      int nn = lane & 15, hi = lane >> 4;
#pragma unroll
      for (int r = 0; r < 8; ++r) {
        int dk = tm + r + 8 * hi, dv = tn + nn;
        float sv = lamL * S[dk][dv] + acc[r];
        S[dk][dv] = sv;
        Sb[dk][dv] = (__bf16)sv;
      }
    }
    __syncthreads();
  }
}

// ---------------------------------------------------------------------------
// gated RMSNorm * silu(z) -> bf16
// ---------------------------------------------------------------------------
__global__ __launch_bounds__(128) void norm_gate_kernel(
    const float* __restrict__ of, const float* __restrict__ zf,
    const float* __restrict__ nw, __bf16* __restrict__ ob) {
  int blk = blockIdx.x;
  int hv = blk % HV_;
  int row = blk / HV_;
  int d = threadIdx.x;
  size_t idx = ((size_t)row * HV_ + hv) * DV_ + d;
  float o = of[idx];
  __shared__ float red[128];
  red[d] = o * o;
  __syncthreads();
#pragma unroll
  for (int s = 64; s > 0; s >>= 1) {
    if (d < s) red[d] += red[d + s];
    __syncthreads();
  }
  float var = red[0] * (1.0f / DV_);
  float on = o * rsqrtf(var + 1e-6f) * nw[d];
  float z = zf[idx];
  ob[idx] = (__bf16)(on * (z / (1.f + __expf(-z))));
}

// ---------------------------------------------------------------------------
// launcher
// ---------------------------------------------------------------------------
static inline size_t align256(size_t x) { return (x + 255) & ~(size_t)255; }

extern "C" void kernel_launch(void* const* d_in, const int* in_sizes, int n_in,
                              void* d_out, int out_size, void* d_ws,
                              size_t ws_size, hipStream_t stream) {
  const float* x       = (const float*)d_in[0];
  const float* W_qkv   = (const float*)d_in[1];
  const float* W_z     = (const float*)d_in[2];
  const float* W_a     = (const float*)d_in[3];
  const float* W_b     = (const float*)d_in[4];
  const float* conv_w  = (const float*)d_in[5];
  const float* A_log   = (const float*)d_in[6];
  const float* dt_bias = (const float*)d_in[7];
  const float* norm_w  = (const float*)d_in[8];
  const float* W_out   = (const float*)d_in[9];

  char* ws = (char*)d_ws;
  size_t off = 0;
  auto alloc = [&](size_t bytes) {
    void* p = ws + off;
    off += align256(bytes);
    return p;
  };

  __bf16* xb    = (__bf16*)alloc((size_t)ROWS_ * DIM_ * 2);
  __bf16* wqkvt = (__bf16*)alloc((size_t)DIM_ * CONVDIM_ * 2); // [K][N]
  __bf16* wzt   = (__bf16*)alloc((size_t)DIM_ * VALDIM_ * 2);
  __bf16* wat   = (__bf16*)alloc((size_t)DIM_ * HV_ * 2);
  __bf16* wbt   = (__bf16*)alloc((size_t)DIM_ * HV_ * 2);
  __bf16* woutt = (__bf16*)alloc((size_t)VALDIM_ * DIM_ * 2);
  float* qkvraw = (float*)alloc((size_t)ROWS_ * CONVDIM_ * 4);
  float* zf     = (float*)alloc((size_t)ROWS_ * VALDIM_ * 4);
  float* af     = (float*)alloc((size_t)ROWS_ * HV_ * 4);
  float* bfp    = (float*)alloc((size_t)ROWS_ * HV_ * 4);
  __bf16* qn    = (__bf16*)alloc((size_t)ROWS_ * KEYDIM_ * 2);
  __bf16* kn    = (__bf16*)alloc((size_t)ROWS_ * KEYDIM_ * 2);
  float* vf     = (float*)alloc((size_t)ROWS_ * VALDIM_ * 4);
  float* gg     = (float*)alloc((size_t)ROWS_ * HV_ * 4);
  float* bb     = (float*)alloc((size_t)ROWS_ * HV_ * 4);
  float* of     = (float*)alloc((size_t)ROWS_ * VALDIM_ * 4);
  __bf16* ob    = (__bf16*)alloc((size_t)ROWS_ * VALDIM_ * 2);

  cvt_bf16_kernel<<<dim3((ROWS_ * DIM_ + 255) / 256), 256, 0, stream>>>(
      x, xb, (size_t)ROWS_ * DIM_);

  auto cvtT = [&](const float* in, __bf16* out, int N, int K) {
    dim3 grid(K / 32, (N + 31) / 32);
    cvtT_kernel<<<grid, 256, 0, stream>>>(in, out, N, K);
  };
  cvtT(W_qkv, wqkvt, CONVDIM_, DIM_);
  cvtT(W_z, wzt, VALDIM_, DIM_);
  cvtT(W_a, wat, HV_, DIM_);
  cvtT(W_b, wbt, HV_, DIM_);
  cvtT(W_out, woutt, DIM_, VALDIM_);

  auto gemm = [&](const __bf16* A, const __bf16* Wt, float* C, int M, int N,
                  int K) {
    dim3 grid((N + GBN - 1) / GBN, M / GBM);
    gemm_bf16_kernel<<<grid, 256, 0, stream>>>(A, Wt, C, M, N, K);
  };
  gemm(xb, wqkvt, qkvraw, ROWS_, CONVDIM_, DIM_);
  gemm(xb, wzt, zf, ROWS_, VALDIM_, DIM_);
  gemm(xb, wat, af, ROWS_, HV_, DIM_);
  gemm(xb, wbt, bfp, ROWS_, HV_, DIM_);

  gates_kernel<<<dim3((ROWS_ * HV_ + 255) / 256), 256, 0, stream>>>(
      af, bfp, A_log, dt_bias, gg, bb, ROWS_ * HV_);

  conv_qk_kernel<<<dim3(B_ * L_ * HK_), 128, 0, stream>>>(qkvraw, conv_w, qn,
                                                          kn);
  conv_v_kernel<<<dim3(B_ * L_ * HV_), 128, 0, stream>>>(qkvraw, conv_w, vf);

  delta_core_kernel<<<dim3(B_ * HV_), 256, 0, stream>>>(qn, kn, vf, gg, bb,
                                                        of);

  norm_gate_kernel<<<dim3(ROWS_ * HV_), 128, 0, stream>>>(of, zf, norm_w, ob);

  gemm(ob, woutt, (float*)d_out, ROWS_, DIM_, VALDIM_);
}
